// MultiHeadAttention_52312701665815
// MI455X (gfx1250) — compile-verified, matrix operand order checked by reading, not run on hardware
//
#include <hip/hip_runtime.h>

// ---------------------------------------------------------------------------
// MultiHeadAttention for MI455X (gfx1250): f16 WMMA pipeline, f32 accumulate.
// B=2, S=2048, D=1024, H=16, Dh=64.
// ---------------------------------------------------------------------------

typedef __attribute__((ext_vector_type(16))) _Float16 v16h;
typedef __attribute__((ext_vector_type(8)))  _Float16 v8h;
typedef __attribute__((ext_vector_type(4)))  _Float16 v4h;
typedef __attribute__((ext_vector_type(8)))  float    v8f;
typedef __attribute__((ext_vector_type(4)))  float    v4f;
typedef __attribute__((ext_vector_type(4)))  int      v4i;

#define WMMA16(a, b, c) \
  __builtin_amdgcn_wmma_f32_16x16x32_f16(false, (a), false, (b), (short)0, (c), false, false)

constexpr int Bc = 2, Sc = 2048, Dc = 1024, Hc = 16, Dh = 64;
constexpr int Mg = Bc * Sc;   // 4096 rows for the projection GEMMs
constexpr int Kg = Dc;        // 1024
constexpr int Ng = Dc;        // 1024

// ---------------------------------------------------------------------------
// Async global->LDS staging (ASYNCcnt path) with sync fallback.
// Builtin signature (from hipcc diagnostic): (v4i addrspace(1)*, v4i addrspace(3)*,
// imm offset, imm cpol).
// ---------------------------------------------------------------------------
#if __has_builtin(__builtin_amdgcn_global_load_async_to_lds_b128)
#define USE_ASYNC_LDS 1
typedef __attribute__((address_space(1))) v4i as1_v4i;
typedef __attribute__((address_space(3))) v4i as3_v4i;
#else
#define USE_ASYNC_LDS 0
#endif

__device__ __forceinline__ void stage16B(const _Float16* g, _Float16* l) {
#if USE_ASYNC_LDS
  __builtin_amdgcn_global_load_async_to_lds_b128((as1_v4i*)(g), (as3_v4i*)(l), 0, 0);
#else
  *(v8h*)l = *(const v8h*)g;
#endif
}

__device__ __forceinline__ void stage_wait() {
#if USE_ASYNC_LDS
#if __has_builtin(__builtin_amdgcn_s_wait_asynccnt)
  __builtin_amdgcn_s_wait_asynccnt(0);
#else
  asm volatile("s_wait_asynccnt 0" ::: "memory");
#endif
#endif
}

// ---------------------------------------------------------------------------
// DPP16 butterfly reductions over the 16 lanes of each half-wave (rows of the
// WMMA C layout live in 16-lane halves). VALU-only: no LDS bpermute traffic.
// ---------------------------------------------------------------------------
#if __has_builtin(__builtin_amdgcn_update_dpp)
#define USE_DPP_RED 1
template <int CTRL>
__device__ __forceinline__ float dpp_movf(float x) {
  return __builtin_bit_cast(
      float, __builtin_amdgcn_update_dpp(0, __builtin_bit_cast(int, x), CTRL,
                                         0xF, 0xF, true));
}
__device__ __forceinline__ float red_max16(float x) {
  x = fmaxf(x, dpp_movf<0xB1>(x));   // quad_perm [1,0,3,2]  (xor 1)
  x = fmaxf(x, dpp_movf<0x4E>(x));   // quad_perm [2,3,0,1]  (xor 2)
  x = fmaxf(x, dpp_movf<0x141>(x));  // row_half_mirror      (other quad)
  x = fmaxf(x, dpp_movf<0x140>(x));  // row_mirror           (other octet)
  return x;
}
__device__ __forceinline__ float red_sum16(float x) {
  x += dpp_movf<0xB1>(x);
  x += dpp_movf<0x4E>(x);
  x += dpp_movf<0x141>(x);
  x += dpp_movf<0x140>(x);
  return x;
}
#else
#define USE_DPP_RED 0
__device__ __forceinline__ float red_max16(float x) {
  x = fmaxf(x, __shfl_xor(x, 1, 32));
  x = fmaxf(x, __shfl_xor(x, 2, 32));
  x = fmaxf(x, __shfl_xor(x, 4, 32));
  x = fmaxf(x, __shfl_xor(x, 8, 32));
  return x;
}
__device__ __forceinline__ float red_sum16(float x) {
  x += __shfl_xor(x, 1, 32);
  x += __shfl_xor(x, 2, 32);
  x += __shfl_xor(x, 4, 32);
  x += __shfl_xor(x, 8, 32);
  return x;
}
#endif

// ---------------------------------------------------------------------------
// Fragment loader: row-major [rows x >=32] halves with row stride `ld` (halves).
// Produces the 16-bit 16x32 A-matrix layout (also used for B as column-owner):
//   lane L (r = L%16, hi = L/16) holds halves k = hi*8 + {0..7} and 16+hi*8+{0..7}
// ---------------------------------------------------------------------------
__device__ __forceinline__ v16h load_frag(const _Float16* src, int ld) {
  const int lane = threadIdx.x & 31;
  const int r = lane & 15, hi = lane >> 4;
  const _Float16* p = src + r * ld + hi * 8;
  union { v16h v; v8h h[2]; } u;
  u.h[0] = *(const v8h*)(p);
  u.h[1] = *(const v8h*)(p + 16);
  return u.v;
}

// ---------------------------------------------------------------------------
// fp32 -> f16 conversion (4 elements / thread, 8B stores)
// ---------------------------------------------------------------------------
__global__ __launch_bounds__(256) void mha_cvt_f16(const float* __restrict__ in,
                                                   _Float16* __restrict__ out, int n) {
  int i = (blockIdx.x * 256 + threadIdx.x) * 4;
  if (i < n) {
    v4f x = *(const v4f*)(in + i);
    v4h y;
    y.x = (_Float16)x.x; y.y = (_Float16)x.y;
    y.z = (_Float16)x.z; y.w = (_Float16)x.w;
    *(v4h*)(out + i) = y;
  }
}

// ---------------------------------------------------------------------------
// Tiled GEMM:  C[m,n] = sum_k A[m,k] * W[n,k] + bias[n]
// Block tile 128x128, K-tile 32. 8 waves, wave tile 32x64 (2x4 WMMA accums).
// PROJ=true : write f16 into [B,H,...] layout via (strideS, strideD)
// PROJ=false: write f32 row-major [Mg x Ng] (final output + bias)
// ---------------------------------------------------------------------------
constexpr int BM = 128, BN = 128, BK = 32;
constexpr int LDT = 40;  // LDS row stride in halves (pad: 80B rows, 16B aligned)

template <bool PROJ>
__global__ __launch_bounds__(256) void mha_gemm_f16(
    const _Float16* __restrict__ A, const _Float16* __restrict__ W,
    const float* __restrict__ bias, _Float16* __restrict__ outH,
    float* __restrict__ outF, int strideS, int strideD) {
  __shared__ _Float16 sA[BM * LDT];
  __shared__ _Float16 sW[BN * LDT];

  const int tid = threadIdx.x;
  const int wave = tid >> 5;
  const int waveM = wave & 3;   // 4 waves along M
  const int waveN = wave >> 2;  // 2 waves along N
  const int m0 = blockIdx.x * BM;
  const int n0 = blockIdx.y * BN;
  const int wm0 = waveM * 32, wn0 = waveN * 64;

  v8f acc[2][4];
#pragma unroll
  for (int i = 0; i < 2; i++)
#pragma unroll
    for (int j = 0; j < 4; j++)
#pragma unroll
      for (int e = 0; e < 8; e++) acc[i][j][e] = 0.0f;

  const int row = tid >> 1;
  const int hoff = (tid & 1) * 16;

  for (int k0 = 0; k0 < Kg; k0 += BK) {
    __syncthreads();
    // Cooperative G->LDS: each thread stages 16 halves of A and of W.
    const _Float16* ga = A + (long)(m0 + row) * Kg + k0 + hoff;
    const _Float16* gw = W + (long)(n0 + row) * Kg + k0 + hoff;
    _Float16* la = sA + row * LDT + hoff;
    _Float16* lw = sW + row * LDT + hoff;
    stage16B(ga, la);
    stage16B(ga + 8, la + 8);
    stage16B(gw, lw);
    stage16B(gw + 8, lw + 8);
    if (k0 + BK < Kg) {  // prefetch next K tile -> global_prefetch_b8
      __builtin_prefetch(ga + BK, 0, 3);
      __builtin_prefetch(gw + BK, 0, 3);
    }
    stage_wait();
    __syncthreads();

    v16h af[2], bf[4];
#pragma unroll
    for (int i = 0; i < 2; i++) af[i] = load_frag(sA + (wm0 + 16 * i) * LDT, LDT);
#pragma unroll
    for (int j = 0; j < 4; j++) bf[j] = load_frag(sW + (wn0 + 16 * j) * LDT, LDT);
#pragma unroll
    for (int i = 0; i < 2; i++)
#pragma unroll
      for (int j = 0; j < 4; j++) acc[i][j] = WMMA16(af[i], bf[j], acc[i][j]);
  }

  // Epilogue. C layout: VGPR g, lane L -> (M = g + 8*(L/16), N = L%16).
  const int lane = tid & 31;
  const int col = lane & 15, hi = lane >> 4;
#pragma unroll
  for (int i = 0; i < 2; i++)
#pragma unroll
    for (int j = 0; j < 4; j++) {
      const int nn = n0 + wn0 + 16 * j + col;
      const float bv = bias[nn];
#pragma unroll
      for (int g = 0; g < 8; g++) {
        const int mm = m0 + wm0 + 16 * i + g + 8 * hi;
        const float val = acc[i][j][g] + bv;
        if (PROJ) {
          const int b = mm >> 11, s = mm & (Sc - 1);   // Sc = 2048
          const int h = nn >> 6, d = nn & (Dh - 1);    // Dh = 64
          const long off = (long)(b * Hc + h) * Sc * Dh +
                           (long)s * strideS + (long)d * strideD;
          outH[off] = (_Float16)val;
        } else {
          outF[(long)mm * Ng + nn] = val;
        }
      }
    }
}

// ---------------------------------------------------------------------------
// Flash attention.  q,k: [B,H,S,Dh] f16.  vt: [B,H,Dh,S] f16 (V transposed).
// Output o: [B,S,D] f16.  Each wave owns a 16-row query tile; block stages
// 32-key K / V^T chunks in LDS shared by all 8 waves. Softmax in exp2 domain.
// ---------------------------------------------------------------------------
__global__ __launch_bounds__(256) void mha_flash_attn(
    const _Float16* __restrict__ q, const _Float16* __restrict__ k,
    const _Float16* __restrict__ vt, _Float16* __restrict__ o) {
  constexpr int LKD = 72;  // K tile row stride (halves): 32 key-rows x 64 d
  constexpr int LVT = 40;  // V^T tile row stride: 64 d-rows x 32 keys
  constexpr int LP  = 40;  // per-wave P buffer: 16 x 32
  __shared__ _Float16 sK[32 * LKD];
  __shared__ _Float16 sVT[64 * LVT];
  __shared__ _Float16 sP[8 * 16 * LP];

  const int tid = threadIdx.x;
  const int wave = tid >> 5;
  const int lane = tid & 31;
  const int col = lane & 15, hi = lane >> 4;
  const int bh = blockIdx.y;                   // b*H + h, 0..31
  const int q0 = blockIdx.x * 128 + wave * 16; // this wave's query rows

  const _Float16* qbase = q + (long)bh * Sc * Dh + (long)q0 * Dh;
  const _Float16* kbase = k + (long)bh * Sc * Dh;
  const _Float16* vbase = vt + (long)bh * Dh * Sc;

  // Q fragments (16 x 64 = two 16x32 A-frags), loaded once from global.
  const v16h aq0 = load_frag(qbase, Dh);
  const v16h aq1 = load_frag(qbase + 32, Dh);

  v8f oacc[4];
#pragma unroll
  for (int j = 0; j < 4; j++)
#pragma unroll
    for (int e = 0; e < 8; e++) oacc[j][e] = 0.0f;
  float mrow[8], lrow[8];
#pragma unroll
  for (int g = 0; g < 8; g++) { mrow[g] = -1e30f; lrow[g] = 0.0f; }

  _Float16* pbuf = sP + wave * 16 * LP;
  // exp2 domain: fold 1/sqrt(Dh) * log2(e) into one scale.
  const float scale2 = 0.125f * 1.44269504088896f;

  // staging coordinates (one 16B chunk per thread per tile)
  const int krow = tid >> 3, kch = (tid & 7) * 8;  // K: 32 rows x 64 halves
  const int vrow = tid >> 2, vch = (tid & 3) * 8;  // VT: 64 rows x 32 halves

  for (int kc = 0; kc < Sc; kc += 32) {
    __syncthreads();
    stage16B(kbase + (long)(kc + krow) * Dh + kch, sK + krow * LKD + kch);
    stage16B(vbase + (long)vrow * Sc + kc + vch, sVT + vrow * LVT + vch);
    stage_wait();
    __syncthreads();

    // Scores: two 16x16 tiles over the 32-key chunk, K-dim = Dh = 64.
    v8f sc[2];
#pragma unroll
    for (int j = 0; j < 2; j++) {
#pragma unroll
      for (int e = 0; e < 8; e++) sc[j][e] = 0.0f;
      const v16h b0 = load_frag(sK + 16 * j * LKD, LKD);       // d 0..31
      const v16h b1 = load_frag(sK + 16 * j * LKD + 32, LKD);  // d 32..63
      sc[j] = WMMA16(aq0, b0, sc[j]);
      sc[j] = WMMA16(aq1, b1, sc[j]);
    }

    // Online softmax (exp2 domain). Row M = g + 8*hi lives in 16 lanes of one
    // half-wave; DPP butterflies reduce it without LDS traffic.
    float alpha[8];
#pragma unroll
    for (int g = 0; g < 8; g++) {
      const float t = red_max16(fmaxf(sc[0][g], sc[1][g]));
      const float nm = fmaxf(mrow[g], t * scale2);
      alpha[g] = exp2f(mrow[g] - nm);
      mrow[g] = nm;
    }
#pragma unroll
    for (int g = 0; g < 8; g++) {
      const float p0 = exp2f(sc[0][g] * scale2 - mrow[g]);
      const float p1 = exp2f(sc[1][g] * scale2 - mrow[g]);
      pbuf[(g + 8 * hi) * LP + col]      = (_Float16)p0;
      pbuf[(g + 8 * hi) * LP + col + 16] = (_Float16)p1;
      lrow[g] = lrow[g] * alpha[g] + red_sum16(p0 + p1);
    }
#pragma unroll
    for (int j = 0; j < 4; j++)
#pragma unroll
      for (int g = 0; g < 8; g++) oacc[j][g] *= alpha[g];

    // O += P(16x32) x V(32x64): A-frag from LDS P, B-frags from V^T tile.
    const v16h ap = load_frag(pbuf, LP);
#pragma unroll
    for (int j = 0; j < 4; j++) {
      const v16h bv = load_frag(sVT + 16 * j * LVT, LVT);
      oacc[j] = WMMA16(ap, bv, oacc[j]);
    }
  }

  // Finalize: divide by l, emit f16 into [B,S,D] for the output projection.
  float inv[8];
#pragma unroll
  for (int g = 0; g < 8; g++) inv[g] = 1.0f / lrow[g];
  const int b = bh >> 4, h = bh & 15;
#pragma unroll
  for (int j = 0; j < 4; j++)
#pragma unroll
    for (int g = 0; g < 8; g++) {
      const int s = q0 + g + 8 * hi;
      const int d = 16 * j + col;
      o[(long)(b * Sc + s) * Dc + h * Dh + d] = (_Float16)(oacc[j][g] * inv[g]);
    }
}

// ---------------------------------------------------------------------------
// Host side
// ---------------------------------------------------------------------------
extern "C" void kernel_launch(void* const* d_in, const int* in_sizes, int n_in,
                              void* d_out, int out_size, void* d_ws, size_t ws_size,
                              hipStream_t stream) {
  (void)in_sizes; (void)n_in; (void)out_size; (void)ws_size;
  const float* Q  = (const float*)d_in[0];
  const float* K  = (const float*)d_in[1];
  const float* V  = (const float*)d_in[2];
  const float* Wq = (const float*)d_in[3];
  const float* bq = (const float*)d_in[4];
  const float* Wk = (const float*)d_in[5];
  const float* bk = (const float*)d_in[6];
  const float* Wv = (const float*)d_in[7];
  const float* bv = (const float*)d_in[8];
  const float* Wo = (const float*)d_in[9];
  const float* bo = (const float*)d_in[10];

  char* ws = (char*)d_ws;
  const size_t MB = 1024 * 1024;
  _Float16* Qh  = (_Float16*)(ws + 0 * MB);   // [Mg,Kg] f16       8 MB
  _Float16* Kh  = (_Float16*)(ws + 8 * MB);   //                   8 MB
  _Float16* Vh  = (_Float16*)(ws + 16 * MB);  //                   8 MB
  _Float16* Wqh = (_Float16*)(ws + 24 * MB);  // [Ng,Kg] f16       2 MB
  _Float16* Wkh = (_Float16*)(ws + 26 * MB);
  _Float16* Wvh = (_Float16*)(ws + 28 * MB);
  _Float16* Woh = (_Float16*)(ws + 30 * MB);
  _Float16* qp  = (_Float16*)(ws + 32 * MB);  // [B,H,S,Dh] f16    8 MB
  _Float16* kp  = (_Float16*)(ws + 40 * MB);  // [B,H,S,Dh]        8 MB
  _Float16* vtp = (_Float16*)(ws + 48 * MB);  // [B,H,Dh,S]        8 MB
  _Float16* oh  = (_Float16*)(ws + 56 * MB);  // [B,S,D]           8 MB

  const int nQ = Bc * Sc * Dc;  // 4194304
  const int nW = Dc * Dc;       // 1048576
  mha_cvt_f16<<<nQ / 1024, 256, 0, stream>>>(Q, Qh, nQ);
  mha_cvt_f16<<<nQ / 1024, 256, 0, stream>>>(K, Kh, nQ);
  mha_cvt_f16<<<nQ / 1024, 256, 0, stream>>>(V, Vh, nQ);
  mha_cvt_f16<<<nW / 1024, 256, 0, stream>>>(Wq, Wqh, nW);
  mha_cvt_f16<<<nW / 1024, 256, 0, stream>>>(Wk, Wkh, nW);
  mha_cvt_f16<<<nW / 1024, 256, 0, stream>>>(Wv, Wvh, nW);
  mha_cvt_f16<<<nW / 1024, 256, 0, stream>>>(Wo, Woh, nW);

  dim3 ggrid(Mg / BM, Ng / BN);  // 32 x 8
  // q,k -> [B,H,S,Dh] (strideS=Dh, strideD=1); v -> [B,H,Dh,S] (strideS=1, strideD=S)
  mha_gemm_f16<true><<<ggrid, 256, 0, stream>>>(Qh, Wqh, bq, qp, nullptr, Dh, 1);
  mha_gemm_f16<true><<<ggrid, 256, 0, stream>>>(Kh, Wkh, bk, kp, nullptr, Dh, 1);
  mha_gemm_f16<true><<<ggrid, 256, 0, stream>>>(Vh, Wvh, bv, vtp, nullptr, 1, Sc);

  mha_flash_attn<<<dim3(Sc / 128, Bc * Hc), 256, 0, stream>>>(qp, kp, vtp, oh);

  mha_gemm_f16<false><<<ggrid, 256, 0, stream>>>(oh, Woh, bo, nullptr,
                                                 (float*)d_out, 0, 0);
}
